// RoPEAttention_34239479284126
// MI455X (gfx1250) — matrix operand order repaired and checked by reading, not compiled, not run on hardware
//
#include <hip/hip_runtime.h>
#include <hip/hip_bf16.h>

#define B_  2
#define L_  2048
#define D_  1024
#define H_  16
#define DK_ 64

typedef _Float16 half_t;
typedef __attribute__((ext_vector_type(16))) _Float16 v16h;
typedef __attribute__((ext_vector_type(8)))  _Float16 v8h;
typedef __attribute__((ext_vector_type(8)))  float    v8f;

union U16H { v16h v; v8h h[2]; };

// Per-lane async copy of 16 bytes global -> LDS (gfx1250 ASYNCcnt path).
// INST_OFFSET applies to both the LDS and global side, so chunk 2 reuses regs.
__device__ __forceinline__ void async_b128(unsigned ldsa, unsigned voff,
                                           unsigned long long sbase) {
    asm volatile("global_load_async_to_lds_b128 %0, %1, %2"
                 :: "v"(ldsa), "v"(voff), "s"(sbase) : "memory");
}
__device__ __forceinline__ void async_b128_off16(unsigned ldsa, unsigned voff,
                                                 unsigned long long sbase) {
    asm volatile("global_load_async_to_lds_b128 %0, %1, %2 offset:16"
                 :: "v"(ldsa), "v"(voff), "s"(sbase) : "memory");
}

// ---------------------------------------------------------------------------
// RoPE on raw q,k (full model dim, rotate_half) + downcast q,k,v to f16
// ---------------------------------------------------------------------------
__global__ void rope_kernel(const float* __restrict__ q, const float* __restrict__ k,
                            const float* __restrict__ v,
                            const float* __restrict__ cs, const float* __restrict__ sn,
                            half_t* __restrict__ qr, half_t* __restrict__ kr,
                            half_t* __restrict__ vh) {
    size_t idx = (size_t)blockIdx.x * blockDim.x + threadIdx.x;
    if (idx >= (size_t)B_ * L_ * D_) return;
    int d = (int)(idx % D_);
    size_t bl = idx / D_;
    int l = (int)(bl % L_);
    float c = cs[(size_t)l * D_ + d];
    float s = sn[(size_t)l * D_ + d];
    float sgn = (d < D_ / 2) ? -1.f : 1.f;
    size_t oidx = (d < D_ / 2) ? idx + D_ / 2 : idx - D_ / 2;
    qr[idx] = (half_t)(q[idx] * c + sgn * q[oidx] * s);
    kr[idx] = (half_t)(k[idx] * c + sgn * k[oidx] * s);
    vh[idx] = (half_t)v[idx];
}

__global__ void cvt_kernel(const float* __restrict__ src, half_t* __restrict__ dst, int n) {
    int i = blockIdx.x * blockDim.x + threadIdx.x;
    if (i < n) dst[i] = (half_t)src[i];
}

// ---------------------------------------------------------------------------
// GEMM: C[M,N] = X[M,K] * W[N,K]^T + bias,  M = B*L = 4096, N = K = D = 1024.
// Block (128 thr, 4 waves) computes a 64(M) x 64(N) tile. The shared 64x32
// B-tile is double-buffer staged into LDS with global_load_async_to_lds_b128
// (ASYNCcnt), consumed as ds_load_b128 B-fragments by all 4 waves.
// MODE 0: store f16 -> [B,H,L,DK]       (Q, K heads)
// MODE 1: store f16 -> [B,H,DK,L]       (V transposed, K-contiguous for PV WMMA)
// MODE 2: store f32 -> [M, D] row-major (final output projection)
// ---------------------------------------------------------------------------
template <int MODE>
__global__ void __launch_bounds__(128)
gemm_kernel(const half_t* __restrict__ X, const half_t* __restrict__ Wt,
            const float* __restrict__ bias,
            half_t* __restrict__ dstH, float* __restrict__ dstF) {
    const int lane = threadIdx.x & 31;
    const int wid  = threadIdx.x >> 5;
    const int t    = threadIdx.x;                   // 0..127
    const int NGB  = D_ / 64;                       // 16 n-group blocks
    const int ng   = blockIdx.x % NGB;
    const int bm   = blockIdx.x / NGB;              // 0..63 (M/64)
    const int hh   = lane >> 4;
    const int nn   = lane & 15;

    // A-fragment row for this wave (m = nn within the wave's 16-row sub-tile)
    const size_t arow = (size_t)(bm * 64 + wid * 16 + nn) * D_;

    // B staging: thread t covers 32 bytes of row srow (64 rows x 64 B = 4 KB)
    __shared__ __align__(16) half_t Bsh[2][64 * 32];
    const int srow = t >> 1;
    const int shalf = (t & 1) * 32;                 // byte offset within row
    const unsigned lds0 = (unsigned)(uintptr_t)(&Bsh[0][0]);
    const unsigned ldsa0 = lds0 + (unsigned)(srow * 64 + shalf);
    const unsigned long long wbase =
        (unsigned long long)(uintptr_t)(Wt + (size_t)(ng * 64) * D_);
    const unsigned voff0 = (unsigned)(srow * (D_ * 2) + shalf);

    // prologue: stage k0 = 0 into buffer 0
    async_b128(ldsa0, voff0, wbase);
    async_b128_off16(ldsa0, voff0, wbase);

    v8f acc0 = {}, acc1 = {}, acc2 = {}, acc3 = {};
    const int NK = D_ / 32;                         // 32 K-steps
    for (int kb = 0; kb < NK; ++kb) {
        const int k0 = kb * 32;
        const int cur = kb & 1;
        if (kb + 1 < NK) {                          // stage next buffer
            const unsigned ldsn = ldsa0 + (unsigned)((cur ^ 1) * 4096);
            const unsigned voffn = voff0 + (unsigned)((k0 + 32) * 2);
            async_b128(ldsn, voffn, wbase);
            async_b128_off16(ldsn, voffn, wbase);
            asm volatile("s_wait_asynccnt 2" ::: "memory");
        } else {
            asm volatile("s_wait_asynccnt 0" ::: "memory");
        }
        __syncthreads();                            // current buffer visible to all

        U16H a;
        a.h[0] = *(const v8h*)(X + arow + k0 + hh * 16);
        a.h[1] = *(const v8h*)(X + arow + k0 + hh * 16 + 8);
        const half_t* Bb = &Bsh[cur][0];
        v16h b0 = *(const v16h*)(Bb + (0 * 16 + nn) * 32 + hh * 16);
        v16h b1 = *(const v16h*)(Bb + (1 * 16 + nn) * 32 + hh * 16);
        v16h b2 = *(const v16h*)(Bb + (2 * 16 + nn) * 32 + hh * 16);
        v16h b3 = *(const v16h*)(Bb + (3 * 16 + nn) * 32 + hh * 16);
        acc0 = __builtin_amdgcn_wmma_f32_16x16x32_f16(false, a.v, false, b0, (short)0, acc0, false, false);
        acc1 = __builtin_amdgcn_wmma_f32_16x16x32_f16(false, a.v, false, b1, (short)0, acc1, false, false);
        acc2 = __builtin_amdgcn_wmma_f32_16x16x32_f16(false, a.v, false, b2, (short)0, acc2, false, false);
        acc3 = __builtin_amdgcn_wmma_f32_16x16x32_f16(false, a.v, false, b3, (short)0, acc3, false, false);
        __syncthreads();                            // done reading before re-stage
    }

#pragma unroll
    for (int j = 0; j < 4; ++j) {
        const v8f& acc = (j == 0) ? acc0 : (j == 1) ? acc1 : (j == 2) ? acc2 : acc3;
        const int C = ng * 64 + j * 16 + nn;
        const float bi = bias[C];
#pragma unroll
        for (int i = 0; i < 8; ++i) {
            const int R = bm * 64 + wid * 16 + i + 8 * hh;   // global row (b*L + l)
            const float val = acc[i] + bi;
            if (MODE == 2) {
                dstF[(size_t)R * D_ + C] = val;
            } else {
                const int b = R / L_, l = R % L_;
                const int h = C / DK_, dk = C % DK_;
                if (MODE == 0)
                    dstH[(((size_t)b * H_ + h) * L_ + l) * DK_ + dk] = (half_t)val;
                else
                    dstH[(((size_t)b * H_ + h) * DK_ + dk) * L_ + l] = (half_t)val;
            }
        }
    }
}

// ---------------------------------------------------------------------------
// Flash-style causal attention. One wave per (b,h, 16-query tile).
// Key blocks of 32: S via 4 WMMAs (DK=64 = 2x32 over two 16-key tiles),
// online softmax (shfl_xor row reductions in 16-lane half-groups),
// P -> wave-private LDS tile (transpose to A-fragment layout), PV via 4 WMMAs.
// ---------------------------------------------------------------------------
__global__ void __launch_bounds__(128)
attn_kernel(const half_t* __restrict__ Qh, const half_t* __restrict__ Kh,
            const half_t* __restrict__ Vt, half_t* __restrict__ ctx) {
    const int lane = threadIdx.x & 31;
    const int wid  = threadIdx.x >> 5;
    const int gw   = blockIdx.x * 4 + wid;          // B*H*(L/16) = 4096 waves
    const int QT   = L_ / 16;
    const int qt   = gw % QT;
    const int bh   = gw / QT;
    const int hh   = lane >> 4;
    const int nn   = lane & 15;
    const int q0   = qt * 16;

    const half_t* Qp = Qh + (size_t)bh * L_ * DK_;
    const half_t* Kp = Kh + (size_t)bh * L_ * DK_;
    const half_t* Vp = Vt + (size_t)bh * DK_ * L_;

    __shared__ __align__(32) half_t Plds[4][16 * 32];
    half_t* Pw = Plds[wid];

    // Q A-fragments for the two DK-halves
    U16H aq0, aq1;
    {
        const half_t* base = Qp + (size_t)(q0 + nn) * DK_;
        aq0.h[0] = *(const v8h*)(base + hh * 16);
        aq0.h[1] = *(const v8h*)(base + hh * 16 + 8);
        aq1.h[0] = *(const v8h*)(base + 32 + hh * 16);
        aq1.h[1] = *(const v8h*)(base + 40 + hh * 16);
    }

    v8f o0 = {}, o1 = {}, o2 = {}, o3 = {};
    float mrun[8], lrun[8];
#pragma unroll
    for (int i = 0; i < 8; ++i) { mrun[i] = -3.0e38f; lrun[i] = 0.f; }

    const int nblk = (q0 + 15) / 32 + 1;
    for (int kb = 0; kb < nblk; ++kb) {
        const int n0 = kb * 32;
        // ---- scores: two 16-key tiles, K=DK split into 2x32
        v8f s1 = {}, s2 = {};
        {
            const size_t r1 = (size_t)(n0 + nn) * DK_ + hh * 16;
            const size_t r2 = (size_t)(n0 + 16 + nn) * DK_ + hh * 16;
            v16h b1a = *(const v16h*)(Kp + r1);
            v16h b1b = *(const v16h*)(Kp + r1 + 32);
            v16h b2a = *(const v16h*)(Kp + r2);
            v16h b2b = *(const v16h*)(Kp + r2 + 32);
            s1 = __builtin_amdgcn_wmma_f32_16x16x32_f16(false, aq0.v, false, b1a, (short)0, s1, false, false);
            s1 = __builtin_amdgcn_wmma_f32_16x16x32_f16(false, aq1.v, false, b1b, (short)0, s1, false, false);
            s2 = __builtin_amdgcn_wmma_f32_16x16x32_f16(false, aq0.v, false, b2a, (short)0, s2, false, false);
            s2 = __builtin_amdgcn_wmma_f32_16x16x32_f16(false, aq1.v, false, b2b, (short)0, s2, false, false);
        }
        // ---- online softmax update per row (row m = i + 8*hh)
#pragma unroll
        for (int i = 0; i < 8; ++i) {
            const int qrow = q0 + i + 8 * hh;
            float v1 = s1[i] * 0.125f;              // 1/sqrt(DK)
            float v2 = s2[i] * 0.125f;
            if (n0 + nn > qrow)      v1 = -1.0e30f;
            if (n0 + 16 + nn > qrow) v2 = -1.0e30f;
            float bm = fmaxf(v1, v2);
#pragma unroll
            for (int x = 1; x < 16; x <<= 1) bm = fmaxf(bm, __shfl_xor(bm, x, 32));
            const float mnew = fmaxf(mrun[i], bm);
            const float p1 = __expf(v1 - mnew);
            const float p2 = __expf(v2 - mnew);
            float rs = p1 + p2;
#pragma unroll
            for (int x = 1; x < 16; x <<= 1) rs += __shfl_xor(rs, x, 32);
            const float sc = __expf(mrun[i] - mnew);
            lrun[i] = lrun[i] * sc + rs;
            mrun[i] = mnew;
            o0[i] *= sc; o1[i] *= sc; o2[i] *= sc; o3[i] *= sc;
            const int m = i + 8 * hh;
            Pw[m * 32 + nn]      = (half_t)p1;
            Pw[m * 32 + 16 + nn] = (half_t)p2;
        }
        asm volatile("s_wait_dscnt 0" ::: "memory");   // LDS stores visible to A-frag loads
        // ---- P A-fragment (16x32, from LDS) and V^T B-fragments, PV WMMAs
        U16H ap;
        ap.h[0] = *(const v8h*)(Pw + nn * 32 + hh * 16);
        ap.h[1] = *(const v8h*)(Pw + nn * 32 + hh * 16 + 8);
        const size_t vb = (size_t)nn * L_ + n0 + hh * 16;
        v16h bv0 = *(const v16h*)(Vp + vb + (size_t)0  * L_);
        v16h bv1 = *(const v16h*)(Vp + vb + (size_t)16 * L_);
        v16h bv2 = *(const v16h*)(Vp + vb + (size_t)32 * L_);
        v16h bv3 = *(const v16h*)(Vp + vb + (size_t)48 * L_);
        o0 = __builtin_amdgcn_wmma_f32_16x16x32_f16(false, ap.v, false, bv0, (short)0, o0, false, false);
        o1 = __builtin_amdgcn_wmma_f32_16x16x32_f16(false, ap.v, false, bv1, (short)0, o1, false, false);
        o2 = __builtin_amdgcn_wmma_f32_16x16x32_f16(false, ap.v, false, bv2, (short)0, o2, false, false);
        o3 = __builtin_amdgcn_wmma_f32_16x16x32_f16(false, ap.v, false, bv3, (short)0, o3, false, false);
    }

    // ---- normalize and scatter ctx back to [B*L, D] f16 row-major
    const int b = bh / H_;
    const int h = bh % H_;
#pragma unroll
    for (int i = 0; i < 8; ++i) {
        const float inv = 1.0f / lrun[i];
        const size_t row = (size_t)b * L_ + q0 + i + 8 * hh;
        const size_t base = row * D_ + (size_t)h * DK_ + nn;
        ctx[base + 0]  = (half_t)(o0[i] * inv);
        ctx[base + 16] = (half_t)(o1[i] * inv);
        ctx[base + 32] = (half_t)(o2[i] * inv);
        ctx[base + 48] = (half_t)(o3[i] * inv);
    }
}

// ---------------------------------------------------------------------------
extern "C" void kernel_launch(void* const* d_in, const int* in_sizes, int n_in,
                              void* d_out, int out_size, void* d_ws, size_t ws_size,
                              hipStream_t stream) {
    const float* q    = (const float*)d_in[0];
    const float* k    = (const float*)d_in[1];
    const float* v    = (const float*)d_in[2];
    const float* cs   = (const float*)d_in[3];
    const float* sn   = (const float*)d_in[4];
    // d_in[5] = mask (bool) -- causality computed analytically
    const float* Wq   = (const float*)d_in[6];
    const float* bq   = (const float*)d_in[7];
    const float* Wk   = (const float*)d_in[8];
    const float* bk   = (const float*)d_in[9];
    const float* Wv   = (const float*)d_in[10];
    const float* bv   = (const float*)d_in[11];
    const float* Wc   = (const float*)d_in[12];
    const float* bc   = (const float*)d_in[13];
    float* out = (float*)d_out;
    (void)in_sizes; (void)n_in; (void)out_size; (void)ws_size;

    const size_t BLD = (size_t)B_ * L_ * D_;   // 4,194,304
    const size_t WW  = (size_t)D_ * D_;        // 1,048,576

    char* w = (char*)d_ws;
    half_t* QR  = (half_t*)w; w += BLD * sizeof(half_t);
    half_t* KR  = (half_t*)w; w += BLD * sizeof(half_t);
    half_t* VH  = (half_t*)w; w += BLD * sizeof(half_t);
    half_t* WQh = (half_t*)w; w += WW * sizeof(half_t);
    half_t* WKh = (half_t*)w; w += WW * sizeof(half_t);
    half_t* WVh = (half_t*)w; w += WW * sizeof(half_t);
    half_t* WCh = (half_t*)w; w += WW * sizeof(half_t);
    half_t* QH  = (half_t*)w; w += BLD * sizeof(half_t);
    half_t* KH  = (half_t*)w; w += BLD * sizeof(half_t);
    half_t* VT  = (half_t*)w; w += BLD * sizeof(half_t);
    half_t* CTX = (half_t*)w; w += BLD * sizeof(half_t);

    // 1) RoPE + downcast
    rope_kernel<<<(int)((BLD + 255) / 256), 256, 0, stream>>>(q, k, v, cs, sn, QR, KR, VH);

    // 2) weight downcasts
    cvt_kernel<<<(int)((WW + 255) / 256), 256, 0, stream>>>(Wq, WQh, (int)WW);
    cvt_kernel<<<(int)((WW + 255) / 256), 256, 0, stream>>>(Wk, WKh, (int)WW);
    cvt_kernel<<<(int)((WW + 255) / 256), 256, 0, stream>>>(Wv, WVh, (int)WW);
    cvt_kernel<<<(int)((WW + 255) / 256), 256, 0, stream>>>(Wc, WCh, (int)WW);

    // 3) projections: (M/64)*(N/64) = 1024 blocks x 128 threads
    gemm_kernel<0><<<1024, 128, 0, stream>>>(QR, WQh, bq, QH, nullptr);
    gemm_kernel<0><<<1024, 128, 0, stream>>>(KR, WKh, bk, KH, nullptr);
    gemm_kernel<1><<<1024, 128, 0, stream>>>(VH, WVh, bv, VT, nullptr);

    // 4) attention: B*H*(L/16) = 4096 waves
    attn_kernel<<<1024, 128, 0, stream>>>(QH, KH, VT, CTX);

    // 5) output projection (f32 out + bias)
    gemm_kernel<2><<<1024, 128, 0, stream>>>(CTX, WCh, bc, nullptr, out);
}
